// VisionMamba_21844203668176
// MI455X (gfx1250) — compile-verified, matrix operand order checked by reading, not run on hardware
//
#include <hip/hip_runtime.h>
#include <hip/hip_bf16.h>
#include <math.h>

// ---------------------------------------------------------------------------
// Vision Mamba forward on MI455X (gfx1250, wave32, WMMA).
// GEMMs: C[M,N] (+)= act(A[M,K] @ W[N,K]^T + bias), f16 operands, f32 acc.
// A and W are pre-converted f16, M/N/K padded so tile loads are branch-free
// 16-byte global_load_b128s matching the CDNA5 16-bit fragment layout.
// ---------------------------------------------------------------------------

typedef __attribute__((ext_vector_type(16))) _Float16 v16h;
typedef __attribute__((ext_vector_type(8)))  _Float16 v8h;
typedef __attribute__((ext_vector_type(8)))  float    v8f;

#define IMG     224
#define PATCH   16
#define CIN     3
#define NCLS    10
#define DMODEL  384
#define NLAYER  12
#define DSTATE  16
#define DTRANK  24
#define DCONV   4
#define DINNER  768
#define BATCH   8
#define GRIDP   14            // IMG/PATCH
#define LTOK    196           // GRIDP*GRIDP
#define NTOK    (BATCH*LTOK)  // 1568 = 49*32
#define EPSV    1e-5f

// ---------------------------------------------------------------------------
// Weight conversion: f32 (N x K) -> f16 (Npad x Kpad), zero-padded.
__global__ void cvt_f16_pad(const float* __restrict__ src, _Float16* __restrict__ dst,
                            int N, int K, int Kpad, int total) {
    int i = blockIdx.x * blockDim.x + threadIdx.x;
    if (i >= total) return;
    int n = i / Kpad, k = i - n * Kpad;
    float v = (n < N && k < K) ? src[(long)n * K + k] : 0.0f;
    dst[i] = (_Float16)v;
}

__global__ void neg_exp_kernel(const float* __restrict__ a, float* __restrict__ out, int n) {
    int i = blockIdx.x * blockDim.x + threadIdx.x;
    if (i < n) out[i] = -__expf(a[i]);
}

// ---------------------------------------------------------------------------
// im2col for stride-16 16x16 patch conv, output f16 GEMM operand.
__global__ void im2col_kernel(const float* __restrict__ x, _Float16* __restrict__ out) {
    long i = (long)blockIdx.x * blockDim.x + threadIdx.x;
    if (i >= (long)NTOK * 768) return;
    int m = (int)(i / 768), col = (int)(i - (long)m * 768);
    int b = m / LTOK, t = m - b * LTOK;
    int py = t / GRIDP, px = t - py * GRIDP;
    int c = col >> 8, rem = col & 255, iy = rem >> 4, ix = rem & 15;
    out[i] = (_Float16)x[(((long)(b * CIN + c) * IMG) + py * PATCH + iy) * IMG + px * PATCH + ix];
}

// ---------------------------------------------------------------------------
// WMMA GEMM: one wave per 32x32 C tile (2x2 register blocking, 4 WMMAs/K-step).
// CDNA5 16-bit fragment: lane(half,idx) row idx holds K = half*8+0..7 (elems
// 0-7) and K = 16+half*8+0..7 (elems 8-15)  =>  two b128 loads per fragment.
// Requirements (guaranteed by caller): A has >= gridDim.x*32 readable rows of
// lda elems; W has >= gridDim.y*32 rows of ldw; Kpad % 32 == 0, lda,ldw >= Kpad,
// all multiples of 8.  Output store is bounds-guarded by (M, N).
__global__ void gemm_wmma_kernel(const _Float16* __restrict__ A, int lda,
                                 const _Float16* __restrict__ W, int ldw,
                                 float* __restrict__ C, int ldc,
                                 _Float16* __restrict__ Ch, int ldch,
                                 int M, int N, int Kpad,
                                 const float* __restrict__ bias,
                                 int act, int accumulate) {
    const int lane = threadIdx.x & 31;
    const int half = lane >> 4;
    const int idx  = lane & 15;
    const int m0 = blockIdx.x * 32;
    const int n0 = blockIdx.y * 32;

    const _Float16* a0 = A + (size_t)(m0 + idx) * lda + half * 8;
    const _Float16* a1 = a0 + (size_t)16 * lda;
    const _Float16* w0 = W + (size_t)(n0 + idx) * ldw + half * 8;
    const _Float16* w1 = w0 + (size_t)16 * ldw;

    v8f acc00 = {}, acc01 = {}, acc10 = {}, acc11 = {};
    for (int k0 = 0; k0 < Kpad; k0 += 32) {
        v8h al0 = *(const v8h*)(a0 + k0);
        v8h ah0 = *(const v8h*)(a0 + k0 + 16);
        v8h al1 = *(const v8h*)(a1 + k0);
        v8h ah1 = *(const v8h*)(a1 + k0 + 16);
        v8h bl0 = *(const v8h*)(w0 + k0);
        v8h bh0 = *(const v8h*)(w0 + k0 + 16);
        v8h bl1 = *(const v8h*)(w1 + k0);
        v8h bh1 = *(const v8h*)(w1 + k0 + 16);
        v16h af0 = __builtin_shufflevector(al0, ah0, 0,1,2,3,4,5,6,7,8,9,10,11,12,13,14,15);
        v16h af1 = __builtin_shufflevector(al1, ah1, 0,1,2,3,4,5,6,7,8,9,10,11,12,13,14,15);
        v16h bf0 = __builtin_shufflevector(bl0, bh0, 0,1,2,3,4,5,6,7,8,9,10,11,12,13,14,15);
        v16h bf1 = __builtin_shufflevector(bl1, bh1, 0,1,2,3,4,5,6,7,8,9,10,11,12,13,14,15);
        acc00 = __builtin_amdgcn_wmma_f32_16x16x32_f16(false, af0, false, bf0, (short)0, acc00, false, false);
        acc01 = __builtin_amdgcn_wmma_f32_16x16x32_f16(false, af0, false, bf1, (short)0, acc01, false, false);
        acc10 = __builtin_amdgcn_wmma_f32_16x16x32_f16(false, af1, false, bf0, (short)0, acc10, false, false);
        acc11 = __builtin_amdgcn_wmma_f32_16x16x32_f16(false, af1, false, bf1, (short)0, acc11, false, false);
    }

    auto store_tile = [&](const v8f& acc, int ti, int tj) {
        int n = n0 + tj * 16 + idx;
        if (n >= N) return;
        float bv = bias ? bias[n] : 0.0f;
#pragma unroll
        for (int r = 0; r < 8; ++r) {
            int m = m0 + ti * 16 + half * 8 + r;
            if (m < M) {
                float v = acc[r] + bv;
                if (act == 1) v = (v > 20.0f) ? v : log1pf(expf(v));   // softplus
                size_t o = (size_t)m * ldc + n;
                if (accumulate) v += C[o];
                C[o] = v;
                if (Ch) Ch[(size_t)m * ldch + n] = (_Float16)v;
            }
        }
    };
    store_tile(acc00, 0, 0);
    store_tile(acc01, 0, 1);
    store_tile(acc10, 1, 0);
    store_tile(acc11, 1, 1);
}

// ---------------------------------------------------------------------------
// RMSNorm over dim 384; one wave32 per token; writes the f16 GEMM operand.
__global__ void rmsnorm_kernel(const float* __restrict__ x, const float* __restrict__ w,
                               _Float16* __restrict__ out, int ntok) {
    int wave = threadIdx.x >> 5, lane = threadIdx.x & 31;
    int tok = blockIdx.x * (blockDim.x >> 5) + wave;
    if (tok >= ntok) return;
    const float* xr = x + (long)tok * DMODEL;
    float s = 0.0f;
#pragma unroll
    for (int i = 0; i < 12; ++i) { float v = xr[lane + 32 * i]; s += v * v; }
#pragma unroll
    for (int o = 16; o > 0; o >>= 1) s += __shfl_xor(s, o, 32);
    float inv = rsqrtf(s * (1.0f / DMODEL) + EPSV);
    _Float16* orow = out + (long)tok * DMODEL;
#pragma unroll
    for (int i = 0; i < 12; ++i) {
        int c = lane + 32 * i;
        orow[c] = (_Float16)(xr[c] * inv * w[c]);
    }
}

// ---------------------------------------------------------------------------
// Causal depthwise conv (d_conv=4, left pad 3) + bias + SiLU.
// Reads from first DINNER cols of xz (stride 2*DINNER); writes f32 (for scan)
// and f16 (x_proj GEMM operand).
__global__ void conv_silu_kernel(const float* __restrict__ xz, const float* __restrict__ cw,
                                 const float* __restrict__ cb,
                                 float* __restrict__ u, _Float16* __restrict__ u16) {
    long i = (long)blockIdx.x * blockDim.x + threadIdx.x;
    if (i >= (long)NTOK * DINNER) return;
    int m = (int)(i / DINNER), d = (int)(i - (long)m * DINNER);
    int b = m / LTOK, l = m - b * LTOK;
    float acc = cb[d];
#pragma unroll
    for (int t = 0; t < DCONV; ++t) {
        int ll = l - (DCONV - 1) + t;
        if (ll >= 0)
            acc += xz[((long)(b * LTOK + ll) * (2 * DINNER)) + d] * cw[d * DCONV + t];
    }
    float s = acc * (1.0f / (1.0f + __expf(-acc)));   // SiLU
    u[i] = s;
    u16[i] = (_Float16)s;
}

// ---------------------------------------------------------------------------
// Selective scan: one thread per (b, d) channel; 16-wide state in registers.
// y = (scan + u*D) * silu(res); writes f16 operand for out_proj GEMM.
__global__ void scan_kernel(const float* __restrict__ delta, const float* __restrict__ u,
                            const float* __restrict__ xdbl, const float* __restrict__ Aneg,
                            const float* __restrict__ Dp, const float* __restrict__ xz,
                            _Float16* __restrict__ y16) {
    int tid = blockIdx.x * blockDim.x + threadIdx.x;
    if (tid >= BATCH * DINNER) return;
    int b = tid / DINNER, d = tid - b * DINNER;
    float h[DSTATE];
    float Ar[DSTATE];
#pragma unroll
    for (int n = 0; n < DSTATE; ++n) { h[n] = 0.0f; Ar[n] = Aneg[d * DSTATE + n]; }
    float Dd = Dp[d];
    for (int l = 0; l < LTOK; ++l) {
        long base = (long)(b * LTOK + l);
        if (l + 1 < LTOK) {   // CDNA5 global_prefetch of next step's rows
            __builtin_prefetch(delta + (base + 1) * DINNER + d, 0, 3);
            __builtin_prefetch(xdbl + (base + 1) * 64 + DTRANK, 0, 3);
        }
        float dl = delta[base * DINNER + d];
        float ul = u[base * DINNER + d];
        const float* bc = xdbl + base * 64;
        float yv = 0.0f;
#pragma unroll
        for (int n = 0; n < DSTATE; ++n) {
            float dA  = __expf(dl * Ar[n]);
            float dBu = dl * bc[DTRANK + n] * ul;
            h[n] = dA * h[n] + dBu;
            yv += h[n] * bc[DTRANK + DSTATE + n];
        }
        yv += ul * Dd;
        float r = xz[base * (2 * DINNER) + DINNER + d];            // res gate
        yv *= r * (1.0f / (1.0f + __expf(-r)));                    // * silu(res)
        y16[base * DINNER + d] = (_Float16)yv;
    }
}

// ---------------------------------------------------------------------------
// Mean pool over L (reads f16 normed activations, writes padded f16 operand;
// rows >= BATCH are zero-filled so the head GEMM reads defined data).
__global__ void meanpool_kernel(const _Float16* __restrict__ hn, _Float16* __restrict__ pooled) {
    int tid = blockIdx.x * blockDim.x + threadIdx.x;
    if (tid >= 32 * DMODEL) return;
    int b = tid / DMODEL, c = tid - b * DMODEL;
    float s = 0.0f;
    if (b < BATCH)
        for (int l = 0; l < LTOK; ++l) s += (float)hn[((long)(b * LTOK + l)) * DMODEL + c];
    pooled[tid] = (_Float16)(s * (1.0f / LTOK));
}

// ---------------------------------------------------------------------------
extern "C" void kernel_launch(void* const* d_in, const int* in_sizes, int n_in,
                              void* d_out, int out_size, void* d_ws, size_t ws_size,
                              hipStream_t stream) {
    (void)in_sizes; (void)n_in; (void)out_size; (void)ws_size;
    const float* x            = (const float*)d_in[0];
    const float* patch_w      = (const float*)d_in[1];
    const float* patch_b      = (const float*)d_in[2];
    const float* in_proj_w    = (const float*)d_in[3];
    const float* conv_w       = (const float*)d_in[4];
    const float* conv_b       = (const float*)d_in[5];
    const float* x_proj_w     = (const float*)d_in[6];
    const float* dt_proj_w    = (const float*)d_in[7];
    const float* dt_proj_b    = (const float*)d_in[8];
    const float* A_log        = (const float*)d_in[9];
    const float* Dvec         = (const float*)d_in[10];
    const float* out_proj_w   = (const float*)d_in[11];
    const float* norm_w       = (const float*)d_in[12];
    const float* final_norm_w = (const float*)d_in[13];
    const float* head_w       = (const float*)d_in[14];
    const float* head_b       = (const float*)d_in[15];
    float* out = (float*)d_out;

    // ---- bump allocator over workspace (~55 MB total) ----
    char* ws = (char*)d_ws;
    size_t off = 0;
    auto alloc = [&](size_t bytes) -> char* {
        char* p = ws + off;
        off += (bytes + 255) & ~(size_t)255;
        return p;
    };
    _Float16* patch16  = (_Float16*)alloc((size_t)DMODEL * 768 * 2);
    _Float16* inproj16 = (_Float16*)alloc((size_t)NLAYER * 2 * DINNER * DMODEL * 2);
    _Float16* xproj16  = (_Float16*)alloc((size_t)NLAYER * 64 * DINNER * 2);     // N 56->64
    _Float16* dtproj16 = (_Float16*)alloc((size_t)NLAYER * DINNER * 32 * 2);     // K 24->32
    _Float16* outp16   = (_Float16*)alloc((size_t)NLAYER * DMODEL * DINNER * 2);
    _Float16* head16   = (_Float16*)alloc((size_t)32 * DMODEL * 2);              // N 10->32
    float*    Aneg     = (float*)alloc((size_t)NLAYER * DINNER * DSTATE * 4);
    float*    h        = (float*)alloc((size_t)NTOK * DMODEL * 4);
    _Float16* hn16     = (_Float16*)alloc((size_t)NTOK * DMODEL * 2);
    _Float16* col16    = (_Float16*)alloc((size_t)NTOK * 768 * 2);
    float*    xz       = (float*)alloc((size_t)NTOK * 2 * DINNER * 4);
    float*    u        = (float*)alloc((size_t)NTOK * DINNER * 4);
    _Float16* u16      = (_Float16*)alloc((size_t)NTOK * DINNER * 2);
    float*    xdbl     = (float*)alloc((size_t)NTOK * 64 * 4);
    _Float16* xdbl16   = (_Float16*)alloc((size_t)NTOK * 64 * 2);
    float*    delta    = (float*)alloc((size_t)NTOK * DINNER * 4);
    _Float16* y16      = (_Float16*)alloc((size_t)NTOK * DINNER * 2);
    _Float16* pooled16 = (_Float16*)alloc((size_t)32 * DMODEL * 2);

    const int TB = 256;
    auto nb = [](long n, int b) { return (unsigned)((n + b - 1) / b); };

    // ---- weight conversion / precompute ----
    cvt_f16_pad<<<nb((long)DMODEL * 768, TB), TB, 0, stream>>>(
        patch_w, patch16, DMODEL, 768, 768, DMODEL * 768);
    cvt_f16_pad<<<nb((long)NLAYER * 2 * DINNER * DMODEL, TB), TB, 0, stream>>>(
        in_proj_w, inproj16, NLAYER * 2 * DINNER, DMODEL, DMODEL, NLAYER * 2 * DINNER * DMODEL);
    cvt_f16_pad<<<nb((long)NLAYER * DMODEL * DINNER, TB), TB, 0, stream>>>(
        out_proj_w, outp16, NLAYER * DMODEL, DINNER, DINNER, NLAYER * DMODEL * DINNER);
    cvt_f16_pad<<<nb((long)32 * DMODEL, TB), TB, 0, stream>>>(
        head_w, head16, NCLS, DMODEL, DMODEL, 32 * DMODEL);
    for (int i = 0; i < NLAYER; ++i) {
        cvt_f16_pad<<<nb((long)64 * DINNER, TB), TB, 0, stream>>>(
            x_proj_w + (long)i * 56 * DINNER, xproj16 + (long)i * 64 * DINNER,
            56, DINNER, DINNER, 64 * DINNER);
        cvt_f16_pad<<<nb((long)DINNER * 32, TB), TB, 0, stream>>>(
            dt_proj_w + (long)i * DINNER * DTRANK, dtproj16 + (long)i * DINNER * 32,
            DINNER, DTRANK, 32, DINNER * 32);
    }
    neg_exp_kernel<<<nb((long)NLAYER * DINNER * DSTATE, TB), TB, 0, stream>>>(
        A_log, Aneg, NLAYER * DINNER * DSTATE);

    // ---- patch embedding: im2col(f16) + WMMA GEMM -> h ----
    im2col_kernel<<<nb((long)NTOK * 768, TB), TB, 0, stream>>>(x, col16);
    gemm_wmma_kernel<<<dim3(NTOK / 32, DMODEL / 32), 32, 0, stream>>>(
        col16, 768, patch16, 768, h, DMODEL, nullptr, 0,
        NTOK, DMODEL, 768, patch_b, 0, 0);

    // ---- 12 mamba layers ----
    for (int i = 0; i < NLAYER; ++i) {
        rmsnorm_kernel<<<NTOK / 8, 256, 0, stream>>>(h, norm_w + (long)i * DMODEL, hn16, NTOK);
        // in_proj: (NTOK,384) x (1536,384)^T -> xz (f32)
        gemm_wmma_kernel<<<dim3(NTOK / 32, (2 * DINNER) / 32), 32, 0, stream>>>(
            hn16, DMODEL, inproj16 + (long)i * 2 * DINNER * DMODEL, DMODEL,
            xz, 2 * DINNER, nullptr, 0, NTOK, 2 * DINNER, DMODEL, nullptr, 0, 0);
        // causal conv + silu -> u (f32) + u16 (f16)
        conv_silu_kernel<<<nb((long)NTOK * DINNER, TB), TB, 0, stream>>>(
            xz, conv_w + (long)i * DINNER * DCONV, conv_b + (long)i * DINNER, u, u16);
        // x_proj: (NTOK,768) x (64,768)^T -> xdbl (f32) + xdbl16 (f16 for dt_proj)
        gemm_wmma_kernel<<<dim3(NTOK / 32, 64 / 32), 32, 0, stream>>>(
            u16, DINNER, xproj16 + (long)i * 64 * DINNER, DINNER,
            xdbl, 64, xdbl16, 64, NTOK, 64, DINNER, nullptr, 0, 0);
        // dt_proj + bias + softplus -> delta (A = cols 0..31 of xdbl16; W cols 24..31 = 0)
        gemm_wmma_kernel<<<dim3(NTOK / 32, DINNER / 32), 32, 0, stream>>>(
            xdbl16, 64, dtproj16 + (long)i * DINNER * 32, 32,
            delta, DINNER, nullptr, 0, NTOK, DINNER, 32,
            dt_proj_b + (long)i * DINNER, 1, 0);
        // selective scan (+ u*D, * silu(res)) -> y16
        scan_kernel<<<nb((long)BATCH * DINNER, TB), TB, 0, stream>>>(
            delta, u, xdbl, Aneg + (long)i * DINNER * DSTATE,
            Dvec + (long)i * DINNER, xz, y16);
        // out_proj with residual accumulate into h
        gemm_wmma_kernel<<<dim3(NTOK / 32, DMODEL / 32), 32, 0, stream>>>(
            y16, DINNER, outp16 + (long)i * DMODEL * DINNER, DINNER,
            h, DMODEL, nullptr, 0, NTOK, DMODEL, DINNER, nullptr, 0, 1);
    }

    // ---- final norm, mean pool, head ----
    rmsnorm_kernel<<<NTOK / 8, 256, 0, stream>>>(h, final_norm_w, hn16, NTOK);
    meanpool_kernel<<<nb((long)32 * DMODEL, TB), TB, 0, stream>>>(hn16, pooled16);
    gemm_wmma_kernel<<<dim3(1, 1), 32, 0, stream>>>(
        pooled16, DMODEL, head16, DMODEL, out, NCLS, nullptr, 0,
        BATCH, NCLS, DMODEL, head_b, 0, 0);
}
